// SymplecticLayer_15814069584136
// MI455X (gfx1250) — matrix operand order repaired and checked by reading, not compiled, not run on hardware
//
#include <hip/hip_runtime.h>
#include <hip/hip_bf16.h>

// ---------------------------------------------------------------------------
// SymplecticLayer fused kernel for gfx1250 (MI455X).
//   GEMMs in bf16 via v_wmma_f32_16x16x32_bf16 (f32 accumulate).
//   One workgroup = 16 rows; z lives in WMMA f32 accumulators across the
//   whole pipeline; LDS used only for bf16 operand tiles, the J-permutation
//   exchange, and the LayerNorm reduction buffer.
//   K-loop is software-pipelined: B tiles for kt+1 are in flight while the
//   WMMAs for kt execute (double-buffered in registers).
// ---------------------------------------------------------------------------

#define HDIM   1024
#define DT_C   0.1f
#define LN_EPS_C 1e-5f
#define ROWS   16          // rows per workgroup (one M tile)
#define WAVES  8           // 256 threads
#define KT     (HDIM / 32) // 32 k-tiles of depth 32
#define NT     8           // 16-wide n-tiles per wave (8*16*8 waves = 1024 cols)

typedef __attribute__((ext_vector_type(16))) __bf16 v16bf;
typedef __attribute__((ext_vector_type(8)))  __bf16 v8bf;
typedef __attribute__((ext_vector_type(8)))  float  v8f;

__device__ __forceinline__ unsigned short f2bf(float f) {
  unsigned int u = __float_as_uint(f);
  u += 0x7fffu + ((u >> 16) & 1u);           // round-to-nearest-even
  return (unsigned short)(u >> 16);
}

// ---------------------------------------------------------------------------
// Pack kernel: builds the three B operands in WMMA-B lane layout.
//   B layout used here (16-bit, 32x16 KxN tile): lane = half*16 + n, the lane's 16
//   contiguous bf16 are k = ktile*32 + half*16 + (0..15)  (contiguous k order, same as A).
//   mat 0: Bz[k][n]   = S_sym[n][k] = (n==k) + S[n][k] - S[k][n]   (z = x @ S_symT)
//   mat 1: Bw1t[k][n] = W1[n][k]                                    (pre = z @ W1T)
//   mat 2: Bw1[k][n]  = W1[k][n]                                    (grad = u @ W1)
// ---------------------------------------------------------------------------
__global__ void symp_pack_kernel(const float* __restrict__ S,
                                 const float* __restrict__ W1,
                                 unsigned short* __restrict__ pk) {
  long long idx = (long long)blockIdx.x * blockDim.x + threadIdx.x;
  const long long per = (long long)HDIM * HDIM;
  if (idx >= 3 * per) return;
  int mat  = (int)(idx / per);
  int e    = (int)(idx % per);
  int j    = e & 15;          // element within lane
  int lane = (e >> 4) & 31;
  int tn   = (e >> 9) & 63;   // 64 n-tiles
  int tk   = e >> 15;         // 32 k-tiles
  int h    = lane >> 4;
  int n    = tn * 16 + (lane & 15);
  int k    = tk * 32 + h * 16 + j;
  float v;
  if (mat == 0)      v = ((n == k) ? 1.0f : 0.0f) + S[n * HDIM + k] - S[k * HDIM + n];
  else if (mat == 1) v = W1[n * HDIM + k];
  else               v = W1[k * HDIM + n];
  pk[idx] = f2bf(v);
}

// ---- operand fetch helpers -------------------------------------------------
__device__ __forceinline__ void load_b_tiles(v16bf (&b)[NT],
                                             const unsigned short* __restrict__ Bpk,
                                             int kt, int w, int lane) {
#pragma unroll
  for (int nt = 0; nt < NT; ++nt)
    b[nt] = *(const v16bf*)(Bpk + (((size_t)(kt * 64 + w * NT + nt)) << 9) + lane * 16);
}

__device__ __forceinline__ v16bf load_a_tile(const unsigned short* __restrict__ Albs,
                                             int kt, int lane) {
  const int h = lane >> 4;
  const int m = lane & 15;
  union { v16bf v; v8bf p[2]; } a;
  const unsigned short* abase = Albs + m * HDIM + kt * 32 + h * 16;
  a.p[0] = *(const v8bf*)(abase);        // k = kt*32 + h*16 + 0..7
  a.p[1] = *(const v8bf*)(abase + 8);    // k = kt*32 + h*16 + 8..15
  return a.v;
}

// ---------------------------------------------------------------------------
// Full K=1024 GEMM accumulation for this wave's 8 n-tiles, software-pipelined:
// while the 8 WMMAs for K-slice kt run, the 16 b128 loads for kt+1 are in
// flight, so the WMMA pipe only waits on the *oldest* outstanding loads.
// ---------------------------------------------------------------------------
__device__ __forceinline__ void gemm_acc(v8f (&acc)[NT],
                                         const unsigned short* __restrict__ Albs,
                                         const unsigned short* __restrict__ Bpk,
                                         int w, int lane) {
  v16bf bA[NT], bB[NT];
  load_b_tiles(bA, Bpk, 0, w, lane);
#pragma unroll 1
  for (int kt = 0; kt < KT; kt += 2) {
    v16bf a0 = load_a_tile(Albs, kt, lane);
    load_b_tiles(bB, Bpk, kt + 1, w, lane);              // in flight during WMMAs
    __builtin_prefetch(Bpk + (((size_t)(((kt + 2) & (KT - 1)) * 64 + w * NT)) << 9) + lane * 16, 0, 1);
#pragma unroll
    for (int nt = 0; nt < NT; ++nt)
      acc[nt] = __builtin_amdgcn_wmma_f32_16x16x32_bf16(
          false, a0, false, bA[nt], (short)0, acc[nt], false, false);
    v16bf a1 = load_a_tile(Albs, kt + 1, lane);
    if (kt + 2 < KT)
      load_b_tiles(bA, Bpk, kt + 2, w, lane);            // in flight during WMMAs
#pragma unroll
    for (int nt = 0; nt < NT; ++nt)
      acc[nt] = __builtin_amdgcn_wmma_f32_16x16x32_bf16(
          false, a1, false, bB[nt], (short)0, acc[nt], false, false);
  }
}

__launch_bounds__(256, 1)
__global__ void symp_main_kernel(const float* __restrict__ x,
                                 const float* __restrict__ b1,
                                 const float* __restrict__ w2,
                                 const float* __restrict__ gamma,
                                 const float* __restrict__ beta,
                                 const int*   __restrict__ num_steps_p,
                                 const unsigned short* __restrict__ Bz,
                                 const unsigned short* __restrict__ Bw1t,
                                 const unsigned short* __restrict__ Bw1,
                                 float* __restrict__ out) {
  extern __shared__ unsigned char smem[];
  unsigned short* zb = (unsigned short*)smem;         // bf16 z (A operand)  32 KB
  unsigned short* ub = zb + ROWS * HDIM;              // bf16 u (A operand)  32 KB
  float*          gf = (float*)(ub + ROWS * HDIM);    // f32 scratch         64 KB

  const int tid  = threadIdx.x;
  const int lane = tid & 31;
  const int w    = tid >> 5;
  const int h    = lane >> 4;
  const int n    = lane & 15;
  const int rowBase = blockIdx.x * ROWS;
  const int ns = num_steps_p[0];

  // ---- stage x tile as bf16 into zb ----------------------------------------
  for (int i = tid; i < ROWS * HDIM; i += 256)
    zb[i] = f2bf(x[(long long)rowBase * HDIM + i]);
  __syncthreads();

  // ---- GEMM 1: z = x @ S_symT ; z stays in registers (f32) -----------------
  v8f zacc[NT];
#pragma unroll
  for (int nt = 0; nt < NT; ++nt) zacc[nt] = {};
  gemm_acc(zacc, zb, Bz, w, lane);
  __syncthreads();                                   // all xb reads done
  // publish bf16 copy of z (C-layout: m = v + 8h, col = lane n of the tile)
#pragma unroll
  for (int nt = 0; nt < NT; ++nt) {
    int col = w * 128 + nt * 16 + n;
#pragma unroll
    for (int v = 0; v < 8; ++v)
      zb[(v + 8 * h) * HDIM + col] = f2bf(zacc[nt][v]);
  }
  __syncthreads();

  // ---- Hamiltonian flow steps ----------------------------------------------
  for (int step = 0; step < ns; ++step) {
    // GEMM 2: pre = z @ W1T, then u = (1 - tanh^2(pre + b1)) * w2
    v8f p[NT];
#pragma unroll
    for (int nt = 0; nt < NT; ++nt) p[nt] = {};
    gemm_acc(p, zb, Bw1t, w, lane);
#pragma unroll
    for (int nt = 0; nt < NT; ++nt) {
      int col = w * 128 + nt * 16 + n;
      float b1v = b1[col];
      float w2v = w2[col];
#pragma unroll
      for (int v = 0; v < 8; ++v) {
        float t = tanhf(p[nt][v] + b1v);
        ub[(v + 8 * h) * HDIM + col] = f2bf((1.0f - t * t) * w2v);
      }
    }
    __syncthreads();

    // GEMM 3: grad = u @ W1  -> gf (f32)
#pragma unroll
    for (int nt = 0; nt < NT; ++nt) p[nt] = {};
    gemm_acc(p, ub, Bw1, w, lane);
#pragma unroll
    for (int nt = 0; nt < NT; ++nt) {
      int col = w * 128 + nt * 16 + n;
#pragma unroll
      for (int v = 0; v < 8; ++v)
        gf[(v + 8 * h) * HDIM + col] = p[nt][v];
    }
    __syncthreads();

    // z += DT * (J @ grad): flow[c<512] = grad[c+512], flow[c>=512] = -grad[c-512]
#pragma unroll
    for (int nt = 0; nt < NT; ++nt) {
      int col = w * 128 + nt * 16 + n;
      int src = (col < 512) ? (col + 512) : (col - 512);
      float sgn = (col < 512) ? DT_C : -DT_C;
#pragma unroll
      for (int v = 0; v < 8; ++v) {
        int m = v + 8 * h;
        zacc[nt][v] += sgn * gf[m * HDIM + src];
        zb[m * HDIM + col] = f2bf(zacc[nt][v]);
      }
    }
    __syncthreads();
  }

  // ---- residual + LayerNorm -------------------------------------------------
#pragma unroll
  for (int nt = 0; nt < NT; ++nt) {
    int col = w * 128 + nt * 16 + n;
#pragma unroll
    for (int v = 0; v < 8; ++v) {
      int m = v + 8 * h;
      gf[m * HDIM + col] = zacc[nt][v] + x[(long long)(rowBase + m) * HDIM + col];
    }
  }
  __syncthreads();

  // each wave normalizes 2 rows; lane reduces 32 strided columns, wave32 shfl tree
#pragma unroll
  for (int rr = 0; rr < 2; ++rr) {
    int r = 2 * w + rr;
    float s = 0.0f, ss = 0.0f;
#pragma unroll
    for (int i = 0; i < HDIM / 32; ++i) {
      float yv = gf[r * HDIM + lane + i * 32];
      s += yv; ss += yv * yv;
    }
#pragma unroll
    for (int o = 16; o > 0; o >>= 1) {
      s  += __shfl_xor(s,  o, 32);
      ss += __shfl_xor(ss, o, 32);
    }
    float mu  = s * (1.0f / HDIM);
    float var = ss * (1.0f / HDIM) - mu * mu;
    float inv = rsqrtf(var + LN_EPS_C);
#pragma unroll
    for (int i = 0; i < HDIM / 32; ++i) {
      int c = lane + i * 32;
      float yv = gf[r * HDIM + c];
      out[(long long)(rowBase + r) * HDIM + c] = (yv - mu) * inv * gamma[c] + beta[c];
    }
  }
}

extern "C" void kernel_launch(void* const* d_in, const int* in_sizes, int n_in,
                              void* d_out, int out_size, void* d_ws, size_t ws_size,
                              hipStream_t stream) {
  (void)n_in; (void)out_size; (void)ws_size;
  const float* x     = (const float*)d_in[0];
  const float* S     = (const float*)d_in[1];
  const float* W1    = (const float*)d_in[2];
  const float* b1    = (const float*)d_in[3];
  const float* w2    = (const float*)d_in[4];
  /* d_in[5] = b2 (additive const, drops out of grad) ; d_in[6] = J (hardcoded) */
  const float* gamma = (const float*)d_in[7];
  const float* beta  = (const float*)d_in[8];
  const int*   nst   = (const int*)d_in[9];

  unsigned short* pk = (unsigned short*)d_ws;              // 3 * 2 MB packed bf16
  const unsigned short* Bz   = pk;
  const unsigned short* Bw1t = pk + (size_t)HDIM * HDIM;
  const unsigned short* Bw1  = pk + 2 * (size_t)HDIM * HDIM;

  const int packTotal = 3 * HDIM * HDIM;
  symp_pack_kernel<<<(packTotal + 255) / 256, 256, 0, stream>>>(S, W1, pk);

  const int nrows  = in_sizes[0] / HDIM;   // 8 * 2048 = 16384
  const int blocks = nrows / ROWS;         // 1024 workgroups
  const size_t shmem = (size_t)ROWS * HDIM * 2   // zb
                     + (size_t)ROWS * HDIM * 2   // ub
                     + (size_t)ROWS * HDIM * 4;  // gf  => 128 KB (< 320 KB WGP LDS)
  (void)hipFuncSetAttribute(reinterpret_cast<const void*>(symp_main_kernel),
                            hipFuncAttributeMaxDynamicSharedMemorySize, (int)shmem);
  symp_main_kernel<<<blocks, 256, shmem, stream>>>(x, b1, w2, gamma, beta, nst,
                                                   Bz, Bw1t, Bw1, (float*)d_out);
}
